// myBadTransfomer_3100966388061
// MI455X (gfx1250) — matrix operand order repaired and checked by reading, not compiled, not run on hardware
//
#include <hip/hip_runtime.h>
#include <hip/hip_bf16.h>
#include <stdint.h>

// ---------------------------------------------------------------------------
// Types for gfx1250 WMMA (wave32): 16x16x32 bf16 -> f32 accumulate
// ---------------------------------------------------------------------------
typedef __attribute__((ext_vector_type(16))) __bf16 bf16x16;
typedef __attribute__((ext_vector_type(8)))  __bf16 bf16x8;
typedef __attribute__((ext_vector_type(8)))  float  f32x8;

__device__ __forceinline__ __bf16 f2bf(float f) {
  union { float f; uint32_t u; } v; v.f = f;
  uint32_t r = (v.u + 0x7FFFu + ((v.u >> 16) & 1u)) >> 16;  // RNE
  uint16_t h = (uint16_t)r;
  __bf16 b;
  __builtin_memcpy(&b, &h, 2);
  return b;
}

// CDNA5 async global->LDS copy (16B per lane), tracked by ASYNCcnt.
// LDS address = low 32 bits of the generic pointer (aperture rule,
// CDNA5 ISA).
__device__ __forceinline__ void async_g2l_b128(uint32_t lds_addr,
                                               unsigned long long gaddr) {
  asm volatile("global_load_async_to_lds_b128 %0, %1, off"
               :: "v"(lds_addr), "v"(gaddr) : "memory");
}
__device__ __forceinline__ void wait_async0() {
  asm volatile("s_wait_asynccnt 0x0" ::: "memory");
}

// ---------------------------------------------------------------------------
// fp32 -> bf16 conversion (grid-stride)
// ---------------------------------------------------------------------------
__global__ __launch_bounds__(256)
void k_f32_to_bf16(const float* __restrict__ in, __bf16* __restrict__ out, size_t n) {
  size_t i = (size_t)blockIdx.x * blockDim.x + threadIdx.x;
  size_t stride = (size_t)gridDim.x * blockDim.x;
  for (; i < n; i += stride) out[i] = f2bf(in[i]);
}

// ---------------------------------------------------------------------------
// NT GEMM: C[M,N] = A[M,K] (row-major, ld=lda) * B[N,K]^T (row-major, ld=ldb)
// Block = 256 threads = 8 waves; block tile 128(M) x 256(N), K-step 32.
// Waves arranged 2(M) x 4(N); wave tile 64x64 => 4x4 WMMA subtiles
// (16 v_wmma per K-step per wave, acc = 128 VGPRs).
// Double-buffered LDS tiles filled with GLOBAL_LOAD_ASYNC_TO_LDS_B128,
// synchronized with s_wait_asynccnt + workgroup barrier.
// OUT_MODE: 0 = f32 direct, 1 = bf16 direct, 2 = bf16 transposed (C^T[N,M])
// Batch via blockIdx.z with element strides bsA/bsB/bsC.
// ---------------------------------------------------------------------------
template <int OUT_MODE>
__global__ __launch_bounds__(256)
void k_gemm_nt_wmma(const __bf16* __restrict__ A, int lda, size_t bsA,
                    const __bf16* __restrict__ Bw, int ldb, size_t bsB,
                    void* __restrict__ Cout, int ldc, size_t bsC,
                    int K) {
  constexpr int PITCH = 80;               // 64B row + 16B pad: 20-bank stride
  constexpr int ATILE = 128 * PITCH;      // 10240 B (128 rows x 32 bf16)
  constexpr int BTILE = 256 * PITCH;      // 20480 B (256 rows x 32 bf16)
  constexpr int BUFSTRIDE = ATILE + BTILE;
  __shared__ __align__(16) char smem[2 * BUFSTRIDE];   // 60 KB of 320 KB LDS

  const int t = threadIdx.x;
  const int lane = t & 31;
  const int wave = t >> 5;
  const int wm = wave >> 2;               // 0..1 : 64-row band
  const int wn = wave & 3;                // 0..3 : 64-col band
  const int fr = lane & 15;               // fragment row (M for A, N for B)
  const int fh = lane >> 4;               // fragment half
  const size_t zb = blockIdx.z;

  const int mBlk = blockIdx.y * 128;
  const int nBlk = blockIdx.x * 256;

  const __bf16* Ab = A + zb * bsA;
  const __bf16* Bb = Bw + zb * bsB;

  const uint32_t smemBase = (uint32_t)(uintptr_t)&smem[0];
  const char* sm = (const char*)&smem[0];

  // Copy one 128x32 A tile (8 KB) + one 256x32 B tile (16 KB) for K-offset kk.
  // 16B chunks: A = 512 (2/thread), B = 1024 (4/thread).
  auto copyTiles = [&](int buf, int kk) {
#pragma unroll
    for (int q = 0; q < 2; ++q) {
      const int lin = q * 256 + t;
      const int row = lin >> 2;
      const int colB = (lin & 3) * 16;
      const unsigned long long ga = (unsigned long long)(uintptr_t)(
          (const char*)(Ab + (size_t)(mBlk + row) * lda + kk) + colB);
      async_g2l_b128(smemBase + buf * BUFSTRIDE + row * PITCH + colB, ga);
    }
#pragma unroll
    for (int q = 0; q < 4; ++q) {
      const int lin = q * 256 + t;
      const int row = lin >> 2;
      const int colB = (lin & 3) * 16;
      const unsigned long long gb = (unsigned long long)(uintptr_t)(
          (const char*)(Bb + (size_t)(nBlk + row) * ldb + kk) + colB);
      async_g2l_b128(smemBase + buf * BUFSTRIDE + ATILE + row * PITCH + colB, gb);
    }
  };

  f32x8 acc[4][4] = {};

  copyTiles(0, 0);
  wait_async0();
  __syncthreads();

  int buf = 0;
  for (int k = 0; k < K; k += 32) {
    const bool more = (k + 32) < K;
    if (more) copyTiles(buf ^ 1, k + 32);

    // B fragment (32x16 KxN): lane fr = column n, K = fh*16..+15 contiguous
    bf16x16 bF[4];
#pragma unroll
    for (int j = 0; j < 4; ++j) {
      const char* p = sm + buf * BUFSTRIDE + ATILE +
                      (wn * 64 + j * 16 + fr) * PITCH + fh * 32;
      bf16x8 lo = *(const bf16x8*)(p);
      bf16x8 hi = *(const bf16x8*)(p + 16);
      bF[j] = __builtin_shufflevector(lo, hi, 0,1,2,3,4,5,6,7,8,9,10,11,12,13,14,15);
    }
    // A fragment (16x32 MxK): lane fr holds K = fh*16..+15 contiguous;
    // each aF is reused by 4 WMMAs, so the compiler can rotate one buffer.
#pragma unroll
    for (int i = 0; i < 4; ++i) {
      const char* p = sm + buf * BUFSTRIDE +
                      (wm * 64 + i * 16 + fr) * PITCH + fh * 32;
      bf16x8 lo = *(const bf16x8*)(p);
      bf16x8 hi = *(const bf16x8*)(p + 16);
      bf16x16 aF = __builtin_shufflevector(lo, hi, 0,1,2,3,4,5,6,7,8,9,10,11,12,13,14,15);
#pragma unroll
      for (int j = 0; j < 4; ++j)
        acc[i][j] = __builtin_amdgcn_wmma_f32_16x16x32_bf16(
            false, aF, false, bF[j], (short)0, acc[i][j], false, false);
    }

    if (more) {
      wait_async0();      // my async copies into buf^1 have landed
      __syncthreads();    // everyone done reading buf & done copying
      buf ^= 1;
    }
  }

  const int m0 = mBlk + wm * 64;
  const int n0 = nBlk + wn * 64;

  // C/D layout: VGPR g, lanes 0-15 -> (M = g,   N = lane)
  //                     lanes 16-31 -> (M = g+8, N = lane-16)
#pragma unroll
  for (int i = 0; i < 4; ++i) {
#pragma unroll
    for (int j = 0; j < 4; ++j) {
      if (OUT_MODE == 0) {
        float* C = (float*)Cout + zb * bsC;
        const size_t col = (size_t)(n0 + j * 16 + fr);
#pragma unroll
        for (int g = 0; g < 8; ++g) {
          const size_t row = (size_t)(m0 + i * 16 + fh * 8 + g);
          C[row * (size_t)ldc + col] = acc[i][j][g];
        }
      } else if (OUT_MODE == 1) {
        __bf16* C = (__bf16*)Cout + zb * bsC;
        const size_t col = (size_t)(n0 + j * 16 + fr);
#pragma unroll
        for (int g = 0; g < 8; ++g) {
          const size_t row = (size_t)(m0 + i * 16 + fh * 8 + g);
          C[row * (size_t)ldc + col] = f2bf(acc[i][j][g]);
        }
      } else {
        // transposed: output row = original column; 8 contiguous bf16 / lane
        __bf16* C = (__bf16*)Cout + zb * bsC;
        const size_t orow = (size_t)(n0 + j * 16 + fr);
        bf16x8 ov;
#pragma unroll
        for (int g = 0; g < 8; ++g) ov[g] = f2bf(acc[i][j][g]);
        *(bf16x8*)(C + orow * (size_t)ldc + (m0 + i * 16 + fh * 8)) = ov;
      }
    }
  }
}

// ---------------------------------------------------------------------------
// Row softmax (fp32 in) -> bf16 out; one block (256 thr) per row of `cols`.
// TR=true writes transposed per-batch: out[b*outBatchStride + c*outLd + d]
// ---------------------------------------------------------------------------
template <bool TR>
__global__ __launch_bounds__(256)
void k_softmax_bf16(const float* __restrict__ in, __bf16* __restrict__ out,
                    int cols, int rowsPerBatch, size_t outBatchStride, int outLd) {
  const int row = blockIdx.x;
  const float* rp = in + (size_t)row * cols;
  __shared__ float red[256];
  const int t = threadIdx.x;

  float m = -3.402823466e38f;
  for (int c = t; c < cols; c += 256) m = fmaxf(m, rp[c]);
  red[t] = m; __syncthreads();
  for (int s = 128; s > 0; s >>= 1) {
    if (t < s) red[t] = fmaxf(red[t], red[t + s]);
    __syncthreads();
  }
  const float M = red[0]; __syncthreads();

  float sum = 0.f;
  for (int c = t; c < cols; c += 256) sum += __expf(rp[c] - M);
  red[t] = sum; __syncthreads();
  for (int s = 128; s > 0; s >>= 1) {
    if (t < s) red[t] += red[t + s];
    __syncthreads();
  }
  const float inv = 1.0f / red[0];

  if (TR) {
    const int b = row / rowsPerBatch;
    const int d = row % rowsPerBatch;
    __bf16* op = out + (size_t)b * outBatchStride + d;
    for (int c = t; c < cols; c += 256)
      op[(size_t)c * outLd] = f2bf(__expf(rp[c] - M) * inv);
  } else {
    __bf16* op = out + (size_t)row * cols;
    for (int c = t; c < cols; c += 256)
      op[c] = f2bf(__expf(rp[c] - M) * inv);
  }
}

// ---------------------------------------------------------------------------
// u = bf16( x + silu(t) )
// ---------------------------------------------------------------------------
__global__ __launch_bounds__(256)
void k_add_silu_bf16(const float* __restrict__ x, const float* __restrict__ tb,
                     __bf16* __restrict__ out, size_t n) {
  size_t i = (size_t)blockIdx.x * blockDim.x + threadIdx.x;
  size_t stride = (size_t)gridDim.x * blockDim.x;
  for (; i < n; i += stride) {
    const float tv = tb[i];
    const float si = tv / (1.f + __expf(-tv));
    out[i] = f2bf(x[i] + si);
  }
}

// ---------------------------------------------------------------------------
// Orchestration
// ---------------------------------------------------------------------------
extern "C" void kernel_launch(void* const* d_in, const int* in_sizes, int n_in,
                              void* d_out, int out_size, void* d_ws, size_t ws_size,
                              hipStream_t stream) {
  (void)in_sizes; (void)n_in; (void)out_size; (void)ws_size;
  constexpr int    Di = 4096;            // D
  constexpr int    Si = 2048;            // S per batch
  constexpr int    SBi = 4096;           // B*S flat rows
  constexpr size_t D = 4096, S = 2048;
  constexpr size_t MAT = D * D;          // 16M elements (== SB*D)

  const float* x  = (const float*)d_in[0];
  const float* W1 = (const float*)d_in[1];
  const float* W2 = (const float*)d_in[2];
  const float* W3 = (const float*)d_in[3];
  const float* W4 = (const float*)d_in[4];
  const float* W5 = (const float*)d_in[5];
  float* out = (float*)d_out;

  char* ws = (char*)d_ws;
  // byte offsets (all 4 KB-aligned); bf16 mats = 32 MB each
  __bf16* x_bf  = (__bf16*)(ws + 0);                        // [SB, D]
  __bf16* Wbuf  = (__bf16*)(ws + 1 * MAT * 2);              // [D, D]   (reused 5x)
  __bf16* xAT   = (__bf16*)(ws + 2 * MAT * 2);              // [D, SB]  (xA^T, batches side by side)
  __bf16* xBT   = (__bf16*)(ws + 3 * MAT * 2);              // [D, SB]
  __bf16* xC    = (__bf16*)(ws + 4 * MAT * 2);              // [SB, D]
  float*  xSq   = (float*)(ws + 5 * MAT * 2);               // [2, D, D] fp32 (128 MB)
  __bf16* xSqT  = (__bf16*)(ws + 5 * MAT * 2 + 2 * MAT * 4);// [2, D, D] bf16 transposed
  // aliases of dead buffers:
  float*  xO    = xSq;                                      // [2, S, D] fp32 (64 MB)
  float*  tbuf  = (float*)(ws + 5 * MAT * 2 + MAT * 4);     // [SB, D] fp32 (2nd half of xSq)
  __bf16* xO_bf = x_bf;                                     // [SB, D]
  __bf16* u_bf  = xAT;                                      // [SB, D]

  const dim3 blk(256);
  const dim3 cblk(256);

  // 1) convert x
  k_f32_to_bf16<<<8192, cblk, 0, stream>>>(x, x_bf, MAT);

  // 2) xA^T = (x W1^T)^T   (bf16 transposed store)
  k_f32_to_bf16<<<8192, cblk, 0, stream>>>(W1, Wbuf, MAT);
  k_gemm_nt_wmma<2><<<dim3(16, 32, 1), blk, 0, stream>>>(
      x_bf, Di, 0, Wbuf, Di, 0, xAT, SBi, 0, Di);

  // 3) xB^T
  k_f32_to_bf16<<<8192, cblk, 0, stream>>>(W2, Wbuf, MAT);
  k_gemm_nt_wmma<2><<<dim3(16, 32, 1), blk, 0, stream>>>(
      x_bf, Di, 0, Wbuf, Di, 0, xBT, SBi, 0, Di);

  // 4) xC (bf16 direct)
  k_f32_to_bf16<<<8192, cblk, 0, stream>>>(W3, Wbuf, MAT);
  k_gemm_nt_wmma<1><<<dim3(16, 32, 1), blk, 0, stream>>>(
      x_bf, Di, 0, Wbuf, Di, 0, xC, Di, 0, Di);

  // 5) xSq[b] = xA^T[b] * xB[b]  (NT: A=xAT(+b*2048 cols), B=xBT rows, K=S)
  k_gemm_nt_wmma<0><<<dim3(16, 32, 2), blk, 0, stream>>>(
      xAT, SBi, S, xBT, SBi, S, xSq, Di, MAT, Si);

  // 6) softmax rows of xSq -> xSqT (bf16, transposed per batch)
  k_softmax_bf16<true><<<8192, cblk, 0, stream>>>(xSq, xSqT, Di, Di, MAT, Di);

  // 7) xO[b] = xC[b] * xSq[b]  == NT with B = xSqT[b]
  k_gemm_nt_wmma<0><<<dim3(16, 16, 2), blk, 0, stream>>>(
      xC, Di, S * D, xSqT, Di, MAT, xO, Di, S * D, Di);

  // 8) softmax rows of xO -> xO_bf
  k_softmax_bf16<false><<<4096, cblk, 0, stream>>>(xO, xO_bf, Di, Di, 0, 0);

  // 9) t = xO_bf * W4^T (fp32)
  k_f32_to_bf16<<<8192, cblk, 0, stream>>>(W4, Wbuf, MAT);
  k_gemm_nt_wmma<0><<<dim3(16, 32, 1), blk, 0, stream>>>(
      xO_bf, Di, 0, Wbuf, Di, 0, tbuf, Di, 0, Di);

  // 10) u = bf16(x + silu(t))
  k_add_silu_bf16<<<8192, cblk, 0, stream>>>(x, tbuf, u_bf, MAT);

  // 11) out = u * W5^T (fp32 to d_out)
  k_f32_to_bf16<<<8192, cblk, 0, stream>>>(W5, Wbuf, MAT);
  k_gemm_nt_wmma<0><<<dim3(16, 32, 1), blk, 0, stream>>>(
      u_bf, Di, 0, Wbuf, Di, 0, out, Di, 0, Di);
}
